// Lru_41300405518550
// MI455X (gfx1250) — matrix operand order repaired and checked by reading, not compiled, not run on hardware
//
#include <hip/hip_runtime.h>

// ---------------------------------------------------------------------------
// LRU layer for MI455X (gfx1250, wave32, WMMA).
//
// Pipeline:
//   1. k_const    : per-channel gamma, a=(gr,gi), and A_chunk = a^64
//   2. k_split    : fp32 -> (bf16 hi, bf16 lo) for x, w_in, w_out
//   3. k_gemm     : s = gamma * (x @ w_in^T + b_in)      [split-bf16 WMMA]
//   4. k_scan_*   : 3-phase chunked complex scan over N  (chunk len 64)
//                   phase3 writes feature already split to bf16 hi/lo
//   5. k_gemm     : out = feature @ w_out^T + b_out      [split-bf16 WMMA]
//
// GEMM staging uses GLOBAL_LOAD_ASYNC_TO_LDS_B128 (ASYNCcnt) with a
// double-buffered LDS pipeline when the builtins exist; otherwise falls back
// to the sync global->VGPR->ds_store path.
// ---------------------------------------------------------------------------

#define NSEQ 2048
#define BB   8
#define DD   1024
#define EE   (2 * DD)        // 2048
#define MM   (NSEQ * BB)     // 16384
#define CHUNK 64
#define NCHUNK (NSEQ / CHUNK) // 32

typedef __attribute__((ext_vector_type(16))) __bf16 bf16x16;
typedef __attribute__((ext_vector_type(8)))  float  f32x8;
typedef __attribute__((ext_vector_type(4)))  int    i32x4;

union FragU { bf16x16 v; i32x4 q[2]; };

#if defined(__AMDGCN__) && __has_builtin(__builtin_amdgcn_global_load_async_to_lds_b128)
#define USE_ASYNC_LDS 1
#if __has_builtin(__builtin_amdgcn_s_wait_asynccnt)
#define WAIT_ASYNC() __builtin_amdgcn_s_wait_asynccnt(0)
#else
#define WAIT_ASYNC() asm volatile("s_wait_asynccnt 0x0" ::: "memory")
#endif
#else
#define USE_ASYNC_LDS 0
#endif

#if USE_ASYNC_LDS
// Builtin signature (from clang diagnostic): param0 = v4i global-AS pointer,
// param1 = v4i LDS-AS pointer, then imm offset, imm cpol.
__device__ __forceinline__ void async_copy_b128(const __bf16* g, __bf16* l) {
    __builtin_amdgcn_global_load_async_to_lds_b128(
        (__attribute__((address_space(1))) i32x4*)(i32x4*)(void*)const_cast<__bf16*>(g),
        (__attribute__((address_space(3))) i32x4*)(i32x4*)(void*)l,
        0, 0);
}
#endif

__device__ __forceinline__ void split_bf16(float v, __bf16& h, __bf16& l) {
    h = (__bf16)v;
    l = (__bf16)(v - (float)h);
}

// ---------------------------------------------------------------------------
// Per-channel constants.
// ---------------------------------------------------------------------------
__global__ void k_const(const float* __restrict__ nu_log,
                        const float* __restrict__ theta_log,
                        const float* __restrict__ gamma_log,
                        float* __restrict__ g_gamma,
                        float* __restrict__ g_gr, float* __restrict__ g_gi,
                        float* __restrict__ g_Ar, float* __restrict__ g_Ai) {
    int d = blockIdx.x * blockDim.x + threadIdx.x;
    if (d >= DD) return;
    float nu = __expf(nu_log[d]);
    float th = __expf(theta_log[d]);
    float gm = __expf(-__expf(gamma_log[d]));
    float ar = nu * __cosf(th);
    float ai = nu * __sinf(th);
    g_gamma[d] = gm;
    g_gr[d] = ar;
    g_gi[d] = ai;
    // A_chunk = a^CHUNK via 6 complex squarings (2^6 == 64 == CHUNK)
    float cr = ar, ci = ai;
#pragma unroll
    for (int i = 0; i < 6; ++i) {
        float t = cr * cr - ci * ci;
        ci = 2.0f * cr * ci;
        cr = t;
    }
    g_Ar[d] = cr;
    g_Ai[d] = ci;
}

// ---------------------------------------------------------------------------
// fp32 -> bf16 hi/lo split (4 elements / thread, n divisible by 4).
// ---------------------------------------------------------------------------
__global__ void k_split(const float* __restrict__ in,
                        __bf16* __restrict__ hi, __bf16* __restrict__ lo,
                        int n) {
    int idx = (blockIdx.x * blockDim.x + threadIdx.x) * 4;
    if (idx >= n) return;
    float4 v = *(const float4*)(in + idx);
    union { __bf16 b[4]; float2 f; } uh, ul;
    split_bf16(v.x, uh.b[0], ul.b[0]);
    split_bf16(v.y, uh.b[1], ul.b[1]);
    split_bf16(v.z, uh.b[2], ul.b[2]);
    split_bf16(v.w, uh.b[3], ul.b[3]);
    *(float2*)(hi + idx) = uh.f;
    *(float2*)(lo + idx) = ul.f;
}

// ---------------------------------------------------------------------------
// Split-precision bf16 GEMM:  C[m,n] = epilogue( sum_k A[m,k] * B[n,k] )
// A: (M, K) row-major as hi/lo bf16,  B: (Ncols, K) row-major as hi/lo bf16.
// WG = 256 threads = 8 waves; block tile 128x64; wave tile 32x32 (2x2 WMMA).
// Per K-step, per tile: hi*hi + hi*lo + lo*hi accumulated into f32 C.
// Epilogue: C = g * (acc + bias[n]), g = gamma[n>>1] or 1.
// ---------------------------------------------------------------------------
__global__ __launch_bounds__(256) void
k_gemm(const __bf16* __restrict__ Ahi, const __bf16* __restrict__ Alo,
       const __bf16* __restrict__ Bhi, const __bf16* __restrict__ Blo,
       float* __restrict__ C, int K, int Nld,
       const float* __restrict__ bias, const float* __restrict__ gammav) {
    const int t    = threadIdx.x;
    const int lane = t & 31;
    const int wave = t >> 5;
    const int wm   = (wave & 3) * 32;   // wave M offset in block tile
    const int wn   = (wave >> 2) * 32;  // wave N offset in block tile
    const int m0   = blockIdx.y * 128;
    const int n0   = blockIdx.x * 64;

    const int r = lane & 15;  // row (A) / col (B) within 16x16 tile
    const int h = lane >> 4;  // half-wave selector (K interleave)

    f32x8 acc[2][2] = {};

    // staging decomposition (row pad +8 bf16 -> 80B stride, conflict-free b128)
    const int brow  = t >> 2;                 // B: 64 rows x 4 chunks
    const int bcc   = (t & 3) * 8;

#if USE_ASYNC_LDS
    // ---------------- async double-buffered pipeline ----------------
    __shared__ alignas(16) __bf16 sA[2][2][128][40];   // [buf][hi/lo][row][k+pad]
    __shared__ alignas(16) __bf16 sB[2][2][64][40];

    const int nk = K >> 5;
    auto stage = [&](int ki, int buf) {
#pragma unroll
        for (int j = 0; j < 2; ++j) {
            int chunk = t * 2 + j;            // 0..511
            int row   = chunk >> 2;           // 0..127
            int cc    = (chunk & 3) * 8;      // k sub-offset in elements
            size_t goff = (size_t)(m0 + row) * K + ki * 32 + cc;
            async_copy_b128(Ahi + goff, &sA[buf][0][row][cc]);
            async_copy_b128(Alo + goff, &sA[buf][1][row][cc]);
        }
        size_t goff = (size_t)(n0 + brow) * K + ki * 32 + bcc;
        async_copy_b128(Bhi + goff, &sB[buf][0][brow][bcc]);
        async_copy_b128(Blo + goff, &sB[buf][1][brow][bcc]);
    };

    stage(0, 0);
    for (int ki = 0; ki < nk; ++ki) {
        const int cur = ki & 1;
        WAIT_ASYNC();          // this wave's copies into buf[cur] landed in LDS
        __syncthreads();       // everyone's copies landed; everyone done reading buf[cur^1]
        if (ki + 1 < nk) stage(ki + 1, cur ^ 1);

        FragU ah[2], al[2], bh[2], bl[2];
#pragma unroll
        for (int i = 0; i < 2; ++i) {
            // A: lanes 0-15 row r hold K[0..7],K[16..23]; lanes 16-31 K[8..15],K[24..31]
            const __bf16* pa = &sA[cur][0][wm + i * 16 + r][h * 8];
            ah[i].q[0] = *(const i32x4*)pa;
            ah[i].q[1] = *(const i32x4*)(pa + 16);
            const __bf16* pl = &sA[cur][1][wm + i * 16 + r][h * 8];
            al[i].q[0] = *(const i32x4*)pl;
            al[i].q[1] = *(const i32x4*)(pl + 16);
        }
#pragma unroll
        for (int j = 0; j < 2; ++j) {
            // B: lanes 0-15 col r hold K[0..7],K[16..23]; lanes 16-31 K[8..15],K[24..31]
            const __bf16* pb = &sB[cur][0][wn + j * 16 + r][h * 8];
            bh[j].q[0] = *(const i32x4*)pb;
            bh[j].q[1] = *(const i32x4*)(pb + 16);
            const __bf16* pq = &sB[cur][1][wn + j * 16 + r][h * 8];
            bl[j].q[0] = *(const i32x4*)pq;
            bl[j].q[1] = *(const i32x4*)(pq + 16);
        }
#pragma unroll
        for (int i = 0; i < 2; ++i)
#pragma unroll
            for (int j = 0; j < 2; ++j) {
                acc[i][j] = __builtin_amdgcn_wmma_f32_16x16x32_bf16(
                    false, ah[i].v, false, bh[j].v, (short)0, acc[i][j], false, false);
                acc[i][j] = __builtin_amdgcn_wmma_f32_16x16x32_bf16(
                    false, ah[i].v, false, bl[j].v, (short)0, acc[i][j], false, false);
                acc[i][j] = __builtin_amdgcn_wmma_f32_16x16x32_bf16(
                    false, al[i].v, false, bh[j].v, (short)0, acc[i][j], false, false);
            }
    }
#else
    // ---------------- fallback: sync staging ----------------
    __shared__ alignas(16) __bf16 sA[2][128][40];
    __shared__ alignas(16) __bf16 sB[2][64][40];

    for (int kk = 0; kk < K; kk += 32) {
#pragma unroll
        for (int j = 0; j < 2; ++j) {
            int chunk = t * 2 + j;
            int row   = chunk >> 2;
            int cc    = (chunk & 3) * 8;
            size_t goff = (size_t)(m0 + row) * K + kk + cc;
            *(i32x4*)&sA[0][row][cc] = *(const i32x4*)(Ahi + goff);
            *(i32x4*)&sA[1][row][cc] = *(const i32x4*)(Alo + goff);
        }
        {
            size_t goff = (size_t)(n0 + brow) * K + kk + bcc;
            *(i32x4*)&sB[0][brow][bcc] = *(const i32x4*)(Bhi + goff);
            *(i32x4*)&sB[1][brow][bcc] = *(const i32x4*)(Blo + goff);
        }
        __syncthreads();

        FragU ah[2], al[2], bh[2], bl[2];
#pragma unroll
        for (int i = 0; i < 2; ++i) {
            const __bf16* pa = &sA[0][wm + i * 16 + r][h * 8];
            ah[i].q[0] = *(const i32x4*)pa;
            ah[i].q[1] = *(const i32x4*)(pa + 16);
            const __bf16* pl = &sA[1][wm + i * 16 + r][h * 8];
            al[i].q[0] = *(const i32x4*)pl;
            al[i].q[1] = *(const i32x4*)(pl + 16);
        }
#pragma unroll
        for (int j = 0; j < 2; ++j) {
            const __bf16* pb = &sB[0][wn + j * 16 + r][h * 8];
            bh[j].q[0] = *(const i32x4*)pb;
            bh[j].q[1] = *(const i32x4*)(pb + 16);
            const __bf16* pq = &sB[1][wn + j * 16 + r][h * 8];
            bl[j].q[0] = *(const i32x4*)pq;
            bl[j].q[1] = *(const i32x4*)(pq + 16);
        }
#pragma unroll
        for (int i = 0; i < 2; ++i)
#pragma unroll
            for (int j = 0; j < 2; ++j) {
                acc[i][j] = __builtin_amdgcn_wmma_f32_16x16x32_bf16(
                    false, ah[i].v, false, bh[j].v, (short)0, acc[i][j], false, false);
                acc[i][j] = __builtin_amdgcn_wmma_f32_16x16x32_bf16(
                    false, ah[i].v, false, bl[j].v, (short)0, acc[i][j], false, false);
                acc[i][j] = __builtin_amdgcn_wmma_f32_16x16x32_bf16(
                    false, al[i].v, false, bh[j].v, (short)0, acc[i][j], false, false);
            }
        __syncthreads();
    }
#endif

    // ---- epilogue: C layout lanes 0-15 -> M=v, lanes 16-31 -> M=v+8; N=lane&15
#pragma unroll
    for (int i = 0; i < 2; ++i)
#pragma unroll
        for (int j = 0; j < 2; ++j) {
            int col = n0 + wn + j * 16 + r;
            float bs = bias[col];
            float g  = gammav ? gammav[col >> 1] : 1.0f;
#pragma unroll
            for (int rr = 0; rr < 8; ++rr) {
                int row = m0 + wm + i * 16 + rr + h * 8;
                C[(size_t)row * Nld + col] = g * (acc[i][j][rr] + bs);
            }
        }
}

// ---------------------------------------------------------------------------
// Scan phase 1: per-chunk input aggregate X_c (running h = a*h + x, h0 = 0).
// One thread per (chunk, b, d).
// ---------------------------------------------------------------------------
__global__ void k_scan_chunk(const float* __restrict__ s,
                             const float* __restrict__ g_gr,
                             const float* __restrict__ g_gi,
                             float2* __restrict__ xagg) {
    int id = blockIdx.x * blockDim.x + threadIdx.x;   // (c, b, d), d fastest
    int d = id & (DD - 1);
    int b = (id >> 10) & (BB - 1);
    int c = id >> 13;
    float ar = g_gr[d], ai = g_gi[d];
    float hr = 0.0f, hq = 0.0f;
    const float* p = s + ((size_t)(c * CHUNK) * BB + b) * EE + 2 * d;
    const size_t step = (size_t)BB * EE;
    for (int tt = 0; tt < CHUNK; ++tt) {
        float2 xv = *(const float2*)p;
        p += step;
        float nr = fmaf(ar, hr, fmaf(-ai, hq, xv.x));
        float nq = fmaf(ar, hq, fmaf( ai, hr, xv.y));
        hr = nr; hq = nq;
    }
    xagg[id] = make_float2(hr, hq);
}

// ---------------------------------------------------------------------------
// Scan phase 2: exclusive scan of chunk aggregates -> per-chunk carry-in.
// One thread per (b, d); NCHUNK sequential combine steps with A = a^CHUNK.
// ---------------------------------------------------------------------------
__global__ void k_scan_carry(const float2* __restrict__ xagg,
                             const float* __restrict__ g_Ar,
                             const float* __restrict__ g_Ai,
                             float2* __restrict__ carry) {
    int bd = blockIdx.x * blockDim.x + threadIdx.x;   // (b, d)
    int d = bd & (DD - 1);
    float Ar = g_Ar[d], Ai = g_Ai[d];
    float cr = 0.0f, ci = 0.0f;
    for (int c = 0; c < NCHUNK; ++c) {
        int idx = c * (BB * DD) + bd;
        carry[idx] = make_float2(cr, ci);
        float2 xv = xagg[idx];
        float nr = fmaf(Ar, cr, fmaf(-Ai, ci, xv.x));
        float nq = fmaf(Ar, ci, fmaf( Ai, cr, xv.y));
        cr = nr; ci = nq;
    }
}

// ---------------------------------------------------------------------------
// Scan phase 3: re-run chunk with real carry-in, emit feature = [hr | hi]
// directly as split bf16 hi/lo (A operand of GEMM2).
// ---------------------------------------------------------------------------
__global__ void k_scan_apply(const float* __restrict__ s,
                             const float2* __restrict__ carry,
                             const float* __restrict__ g_gr,
                             const float* __restrict__ g_gi,
                             __bf16* __restrict__ fhi,
                             __bf16* __restrict__ flo) {
    int id = blockIdx.x * blockDim.x + threadIdx.x;   // (c, b, d)
    int d = id & (DD - 1);
    int b = (id >> 10) & (BB - 1);
    int c = id >> 13;
    float ar = g_gr[d], ai = g_gi[d];
    float2 h0 = carry[id];
    float hr = h0.x, hq = h0.y;
    const float* p = s + ((size_t)(c * CHUNK) * BB + b) * EE + 2 * d;
    const size_t step = (size_t)BB * EE;
    for (int tt = 0; tt < CHUNK; ++tt) {
        float2 xv = *(const float2*)p;
        p += step;
        float nr = fmaf(ar, hr, fmaf(-ai, hq, xv.x));
        float nq = fmaf(ar, hq, fmaf( ai, hr, xv.y));
        hr = nr; hq = nq;
        size_t m = (size_t)(c * CHUNK + tt) * BB + b;
        __bf16 vh, vl;
        split_bf16(hr, vh, vl);
        fhi[m * EE + d] = vh;
        flo[m * EE + d] = vl;
        split_bf16(hq, vh, vl);
        fhi[m * EE + DD + d] = vh;
        flo[m * EE + DD + d] = vl;
    }
}

// ---------------------------------------------------------------------------
// Launch.
// ---------------------------------------------------------------------------
extern "C" void kernel_launch(void* const* d_in, const int* in_sizes, int n_in,
                              void* d_out, int out_size, void* d_ws, size_t ws_size,
                              hipStream_t stream) {
    const float* x         = (const float*)d_in[0];
    const float* nu_log    = (const float*)d_in[1];
    const float* theta_log = (const float*)d_in[2];
    const float* gamma_log = (const float*)d_in[3];
    const float* w_in      = (const float*)d_in[4];
    const float* b_in      = (const float*)d_in[5];
    const float* w_out     = (const float*)d_in[6];
    const float* b_out     = (const float*)d_in[7];
    float* out = (float*)d_out;

    // bump allocator over workspace (~356 MB used)
    char* ws = (char*)d_ws;
    auto alloc = [&](size_t bytes) -> char* {
        char* p = ws;
        ws += (bytes + 255) & ~(size_t)255;
        return p;
    };
    float*  g_gamma = (float*)alloc(DD * 4);
    float*  g_gr    = (float*)alloc(DD * 4);
    float*  g_gi    = (float*)alloc(DD * 4);
    float*  g_Ar    = (float*)alloc(DD * 4);
    float*  g_Ai    = (float*)alloc(DD * 4);
    __bf16* w_in_hi  = (__bf16*)alloc((size_t)EE * DD * 2);
    __bf16* w_in_lo  = (__bf16*)alloc((size_t)EE * DD * 2);
    __bf16* w_out_hi = (__bf16*)alloc((size_t)DD * EE * 2);
    __bf16* w_out_lo = (__bf16*)alloc((size_t)DD * EE * 2);
    __bf16* x_hi     = (__bf16*)alloc((size_t)MM * DD * 2);
    __bf16* x_lo     = (__bf16*)alloc((size_t)MM * DD * 2);
    float*  s_buf    = (float*)alloc((size_t)MM * EE * 4);
    float2* xagg     = (float2*)alloc((size_t)NCHUNK * BB * DD * 8);
    float2* carry    = (float2*)alloc((size_t)NCHUNK * BB * DD * 8);
    __bf16* f_hi     = (__bf16*)alloc((size_t)MM * EE * 2);
    __bf16* f_lo     = (__bf16*)alloc((size_t)MM * EE * 2);

    // 1. per-channel constants
    k_const<<<DD / 256, 256, 0, stream>>>(nu_log, theta_log, gamma_log,
                                          g_gamma, g_gr, g_gi, g_Ar, g_Ai);
    // 2. fp32 -> bf16 hi/lo splits
    k_split<<<(EE * DD) / 1024, 256, 0, stream>>>(w_in, w_in_hi, w_in_lo, EE * DD);
    k_split<<<(DD * EE) / 1024, 256, 0, stream>>>(w_out, w_out_hi, w_out_lo, DD * EE);
    k_split<<<(MM * DD) / 1024, 256, 0, stream>>>(x, x_hi, x_lo, MM * DD);
    // 3. in_proj GEMM, fused bias + gamma
    dim3 g1(EE / 64, MM / 128);
    k_gemm<<<g1, 256, 0, stream>>>(x_hi, x_lo, w_in_hi, w_in_lo, s_buf,
                                   DD, EE, b_in, g_gamma);
    // 4. chunked complex scan
    k_scan_chunk<<<(NCHUNK * BB * DD) / 256, 256, 0, stream>>>(s_buf, g_gr, g_gi, xagg);
    k_scan_carry<<<(BB * DD) / 256, 256, 0, stream>>>(xagg, g_Ar, g_Ai, carry);
    k_scan_apply<<<(NCHUNK * BB * DD) / 256, 256, 0, stream>>>(s_buf, carry, g_gr, g_gi,
                                                               f_hi, f_lo);
    // 5. out_proj GEMM, fused bias
    dim3 g2(DD / 64, MM / 128);
    k_gemm<<<g2, 256, 0, stream>>>(f_hi, f_lo, w_out_hi, w_out_lo, out,
                                   EE, DD, b_out, nullptr);
}
